// WeiDecoder_1881195675886
// MI455X (gfx1250) — matrix operand order repaired and checked by reading, not compiled, not run on hardware
//
#include <hip/hip_runtime.h>
#include <stdint.h>

// ---------------- problem constants ----------------
#define BB   128
#define TT   1500
#define LL   300
#define VV   1000
#define EE   256
#define HH   128
#define G4   512          // 4*H gate width
#define KL   512          // gates GEMM K = E + H + H
#define KO   256          // logits GEMM K = 2*H
#define NTW  32           // gate n-tiles (512/16)
#define KTW  16           // gate k-tiles (512/32)
#define NTE  63           // logits n-tiles (pad 1000 -> 1008)
#define KTE  8            // logits k-tiles (256/32)
#define NEG_INF_F (-1.0e9f)

// ---------------- WMMA types ----------------
typedef __bf16 v16bf __attribute__((ext_vector_type(16)));
typedef float  v8f   __attribute__((ext_vector_type(8)));

union FragBF {
    v16bf    v;
    uint32_t u[8];
    uint4    q[2];
};

// ---------------- helpers ----------------
__device__ __forceinline__ uint16_t f2bf(float f) {
    uint32_t u = __builtin_bit_cast(uint32_t, f);
    uint32_t r = u + 0x7FFFu + ((u >> 16) & 1u);   // round-to-nearest-even
    return (uint16_t)(r >> 16);
}
__device__ __forceinline__ float bf2f(uint16_t h) {
    uint32_t u = ((uint32_t)h) << 16;
    return __builtin_bit_cast(float, u);
}
__device__ __forceinline__ float sigmoidf_(float x) {
    return 1.0f / (1.0f + __expf(-x));
}
__device__ __forceinline__ float tanhf_(float x) {
    x = fminf(fmaxf(x, -20.0f), 20.0f);
    float e = __expf(2.0f * x);
    return (e - 1.0f) / (e + 1.0f);
}

// ---------------- prep kernels ----------------
// fp32 -> bf16 bulk convert (vector of 4 per thread)
__global__ __launch_bounds__(256) void cvt_bf16_kernel(const float* __restrict__ src,
                                                       uint16_t* __restrict__ dst,
                                                       long long n4) {
    long long i = (long long)blockIdx.x * blockDim.x + threadIdx.x;
    long long stride = (long long)gridDim.x * blockDim.x;
    for (; i < n4; i += stride) {
        float4 f = ((const float4*)src)[i];
        uint2 o;
        o.x = (uint32_t)f2bf(f.x) | ((uint32_t)f2bf(f.y) << 16);
        o.y = (uint32_t)f2bf(f.z) | ((uint32_t)f2bf(f.w) << 16);
        ((uint2*)dst)[i] = o;
    }
}

// Pack combined LSTM weight [W_ih | W_hh] (512x512, row=n, col=k) into
// B-fragment order: frag(ntile,kt)[lane][i], n = ntile*16 + lane%16,
// k = kt*32 + (lane/16)*16 + i
__global__ __launch_bounds__(256) void pack_w_kernel(const float* __restrict__ W_ih,
                                                     const float* __restrict__ W_hh,
                                                     uint16_t* __restrict__ wpk) {
    int total = NTW * KTW * 32 * 16;  // 262144
    for (int idx = blockIdx.x * blockDim.x + threadIdx.x; idx < total;
         idx += gridDim.x * blockDim.x) {
        int i    = idx & 15;
        int lane = (idx >> 4) & 31;
        int kt   = (idx >> 9) & 15;
        int nt   = idx >> 13;
        int n = nt * 16 + (lane & 15);
        int k = kt * 32 + (lane >> 4) * 16 + i;
        float v = (k < 384) ? W_ih[n * 384 + k] : W_hh[n * 128 + (k - 384)];
        wpk[idx] = f2bf(v);
    }
}

// Pack emb^T (K=256 x N=1008 padded) into B-fragment order for logits GEMM.
__global__ __launch_bounds__(256) void pack_emb_kernel(const float* __restrict__ emb,
                                                       uint16_t* __restrict__ epk) {
    int total = NTE * KTE * 32 * 16;  // 258048
    for (int idx = blockIdx.x * blockDim.x + threadIdx.x; idx < total;
         idx += gridDim.x * blockDim.x) {
        int i    = idx & 15;
        int lane = (idx >> 4) & 31;
        int kt   = (idx >> 9) & 7;
        int nt   = idx >> 12;
        int n = nt * 16 + (lane & 15);
        int k = kt * 32 + (lane >> 4) * 16 + i;
        float v = (n < VV) ? emb[n * EE + k] : 0.0f;
        epk[idx] = f2bf(v);
    }
}

// h = c = 0, ctx = value[:,0,:]
__global__ __launch_bounds__(256) void init_state_kernel(const float* __restrict__ value,
                                                         float* __restrict__ h,
                                                         float* __restrict__ c,
                                                         float* __restrict__ ctx) {
    int i = blockIdx.x * blockDim.x + threadIdx.x;
    if (i < BB * HH) {
        int b = i >> 7, d = i & 127;
        h[i] = 0.0f;
        c[i] = 0.0f;
        ctx[i] = value[(long long)b * TT * HH + d];
    }
}

// ---------------- step kernel 1: gates GEMM (WMMA) + LSTM pointwise ----------------
// grid = 8 blocks (16 batch rows each), 512 threads = 16 waves.
__global__ __launch_bounds__(512) void lstm_gates_kernel(
    const float* __restrict__ emb, const int* __restrict__ y,
    const uint16_t* __restrict__ wpk,
    const float* __restrict__ b_ih, const float* __restrict__ b_hh,
    float* __restrict__ h, float* __restrict__ c,
    const float* __restrict__ ctx, int step) {
    __shared__ uint16_t sA[16 * KL];  // 16 KB  A tile (bf16)
    __shared__ float    sG[16 * G4];  // 32 KB  gates (fp32)

    const int tid = threadIdx.x;
    const int bm  = blockIdx.x * 16;

    // stage A = [emb[y[:,t-1]] | ctx | h] as bf16, row-major [m][512]
    for (int e = tid; e < 16 * KL; e += 512) {
        int m = e >> 9, k = e & 511;
        int b = bm + m;
        float v;
        if (k < EE) {
            if (step == 0) v = 0.0f;
            else {
                int tok = y[b * LL + step - 1];
                v = emb[tok * EE + k];
            }
        } else if (k < EE + HH) {
            v = ctx[b * HH + (k - EE)];
        } else {
            v = h[b * HH + (k - EE - HH)];
        }
        sA[e] = f2bf(v);
    }
    __syncthreads();

    const int wv = tid >> 5, ln = tid & 31;
    const int lm = ln & 15, lh = ln >> 4;
    const int nt0 = wv * 2, nt1 = wv * 2 + 1;

    v8f acc0 = {}, acc1 = {};
    const uint32_t* sA32 = (const uint32_t*)sA;

    for (int kt = 0; kt < KTW; ++kt) {
        FragBF a, bf0, bf1;
#pragma unroll
        for (int j = 0; j < 8; ++j) {
            int klocal = lh * 16 + j * 2;
            a.u[j] = sA32[(lm * KL + kt * 32 + klocal) >> 1];
        }
        const uint4* w0 = (const uint4*)(wpk + ((nt0 * KTW + kt) * 32 + ln) * 16);
        bf0.q[0] = w0[0]; bf0.q[1] = w0[1];
        const uint4* w1 = (const uint4*)(wpk + ((nt1 * KTW + kt) * 32 + ln) * 16);
        bf1.q[0] = w1[0]; bf1.q[1] = w1[1];
        acc0 = __builtin_amdgcn_wmma_f32_16x16x32_bf16(false, a.v, false, bf0.v,
                                                       (short)0, acc0, false, false);
        acc1 = __builtin_amdgcn_wmma_f32_16x16x32_bf16(false, a.v, false, bf1.v,
                                                       (short)0, acc1, false, false);
    }

    const int n0 = nt0 * 16 + lm, n1 = nt1 * 16 + lm;
    const float bias0 = b_ih[n0] + b_hh[n0];
    const float bias1 = b_ih[n1] + b_hh[n1];
#pragma unroll
    for (int r = 0; r < 8; ++r) {
        int m = lh * 8 + r;
        sG[m * G4 + n0] = acc0[r] + bias0;
        sG[m * G4 + n1] = acc1[r] + bias1;
    }
    __syncthreads();

    // pointwise LSTM: gate order i | f | g | o
    for (int e = tid; e < 16 * HH; e += 512) {
        int m = e >> 7, d = e & 127;
        int b = bm + m;
        float ig = sigmoidf_(sG[m * G4 + d]);
        float fg = sigmoidf_(sG[m * G4 + 128 + d]);
        float gg = tanhf_  (sG[m * G4 + 256 + d]);
        float og = sigmoidf_(sG[m * G4 + 384 + d]);
        float cn = fg * c[b * HH + d] + ig * gg;
        float hn = og * tanhf_(cn);
        c[b * HH + d] = cn;
        h[b * HH + d] = hn;
    }
}

// ---------------- step kernel 2: attention (bf16 streaming) ----------------
// grid = 128 blocks (one per batch), 256 threads.
__global__ __launch_bounds__(256) void attention_kernel(
    const uint16_t* __restrict__ kbf, const uint16_t* __restrict__ vbf,
    const int* __restrict__ enc_len, const float* __restrict__ h,
    float* __restrict__ ctx, float* __restrict__ attn_out, int step) {
    __shared__ float sh[HH];
    __shared__ float se[TT];
    __shared__ float red[256];

    const int b = blockIdx.x, tid = threadIdx.x;
    if (tid < HH) sh[tid] = h[b * HH + tid];
    __syncthreads();

    const int len = enc_len[b];

    // energy + local max
    float lmax = NEG_INF_F;
    for (int t = tid; t < TT; t += 256) {
        float e;
        if (t >= len) {
            e = NEG_INF_F;
        } else {
            const uint4* kr = (const uint4*)(kbf + ((long long)b * TT + t) * HH);
            float acc = 0.0f;
#pragma unroll
            for (int q = 0; q < 16; ++q) {
                uint4 pk = kr[q];
                int base = q * 8;
                acc = fmaf(bf2f((uint16_t)pk.x),         sh[base + 0], acc);
                acc = fmaf(bf2f((uint16_t)(pk.x >> 16)), sh[base + 1], acc);
                acc = fmaf(bf2f((uint16_t)pk.y),         sh[base + 2], acc);
                acc = fmaf(bf2f((uint16_t)(pk.y >> 16)), sh[base + 3], acc);
                acc = fmaf(bf2f((uint16_t)pk.z),         sh[base + 4], acc);
                acc = fmaf(bf2f((uint16_t)(pk.z >> 16)), sh[base + 5], acc);
                acc = fmaf(bf2f((uint16_t)pk.w),         sh[base + 6], acc);
                acc = fmaf(bf2f((uint16_t)(pk.w >> 16)), sh[base + 7], acc);
            }
            e = acc;
        }
        se[t] = e;
        lmax = fmaxf(lmax, e);
    }
    // block max reduce
    red[tid] = lmax;
    __syncthreads();
    for (int s = 128; s > 0; s >>= 1) {
        if (tid < s) red[tid] = fmaxf(red[tid], red[tid + s]);
        __syncthreads();
    }
    const float mx = red[0];
    __syncthreads();

    // softmax numerator + local sum
    float lsum = 0.0f;
    for (int t = tid; t < TT; t += 256) {
        float p = (t < len) ? __expf(se[t] - mx) : 0.0f;
        se[t] = p;
        lsum += p;
    }
    red[tid] = lsum;
    __syncthreads();
    for (int s = 128; s > 0; s >>= 1) {
        if (tid < s) red[tid] += red[tid + s];
        __syncthreads();
    }
    const float rs = 1.0f / red[0];
    __syncthreads();

    // ctx = (sum_t p[t] * value[b,t,:]) * rs ; split t over two half-groups
    const int d = tid & 127, hf = tid >> 7;
    float acc = 0.0f;
    for (int t = hf; t < TT; t += 2) {
        acc = fmaf(se[t], bf2f(vbf[((long long)b * TT + t) * HH + d]), acc);
    }
    red[tid] = acc;
    __syncthreads();
    if (tid < HH) ctx[b * HH + d] = (red[tid] + red[tid + 128]) * rs;

    if (b == 0) {
        for (int t = tid; t < TT; t += 256)
            attn_out[(long long)step * TT + t] = se[t] * rs;
    }
}

// ---------------- step kernel 3: logits GEMM (WMMA) ----------------
// grid = (8 m-tiles, 4 n-groups), 256 threads = 8 waves, 2 n-tiles each.
__global__ __launch_bounds__(256) void logits_kernel(
    const float* __restrict__ h, const float* __restrict__ ctx,
    const uint16_t* __restrict__ epk, const float* __restrict__ b_out,
    float* __restrict__ pred, int step) {
    __shared__ uint16_t sA[16 * KO];  // 8 KB

    const int tid = threadIdx.x;
    const int mt  = blockIdx.x;  // 0..7
    const int ng  = blockIdx.y;  // 0..3

    for (int e = tid; e < 16 * KO; e += 256) {
        int m = e >> 8, k = e & 255;
        int b = mt * 16 + m;
        float v = (k < HH) ? h[b * HH + k] : ctx[b * HH + (k - HH)];
        sA[e] = f2bf(v);
    }
    __syncthreads();

    const int wv = tid >> 5, ln = tid & 31;
    const int lm = ln & 15, lh = ln >> 4;
    const int nt0 = ng * 16 + wv * 2;
    const int nt1 = nt0 + 1;
    const int nt1c = (nt1 < NTE) ? nt1 : (NTE - 1);  // clamp load, store guarded

    v8f acc0 = {}, acc1 = {};
    const uint32_t* sA32 = (const uint32_t*)sA;

    for (int kt = 0; kt < KTE; ++kt) {
        FragBF a, bf0, bf1;
#pragma unroll
        for (int j = 0; j < 8; ++j) {
            int klocal = lh * 16 + j * 2;
            a.u[j] = sA32[(lm * KO + kt * 32 + klocal) >> 1];
        }
        const uint4* e0 = (const uint4*)(epk + ((nt0 * KTE + kt) * 32 + ln) * 16);
        bf0.q[0] = e0[0]; bf0.q[1] = e0[1];
        const uint4* e1 = (const uint4*)(epk + ((nt1c * KTE + kt) * 32 + ln) * 16);
        bf1.q[0] = e1[0]; bf1.q[1] = e1[1];
        acc0 = __builtin_amdgcn_wmma_f32_16x16x32_bf16(false, a.v, false, bf0.v,
                                                       (short)0, acc0, false, false);
        acc1 = __builtin_amdgcn_wmma_f32_16x16x32_bf16(false, a.v, false, bf1.v,
                                                       (short)0, acc1, false, false);
    }

    const int n0 = nt0 * 16 + lm, n1 = nt1 * 16 + lm;
    const float bo0 = (n0 < VV) ? b_out[n0] : 0.0f;
    const float bo1 = (n1 < VV && nt1 < NTE) ? b_out[n1] : 0.0f;
#pragma unroll
    for (int r = 0; r < 8; ++r) {
        int m = lh * 8 + r;
        int b = mt * 16 + m;
        long long rowoff = ((long long)b * LL + step) * VV;
        if (n0 < VV) pred[rowoff + n0] = acc0[r] + bo0;
        if (n1 < VV && nt1 < NTE) pred[rowoff + n1] = acc1[r] + bo1;
    }
}

// ---------------- host launch ----------------
extern "C" void kernel_launch(void* const* d_in, const int* in_sizes, int n_in,
                              void* d_out, int out_size, void* d_ws, size_t ws_size,
                              hipStream_t stream) {
    const float* key    = (const float*)d_in[0];  // (B,T,H)
    const float* value  = (const float*)d_in[1];  // (B,T,H)
    const int*   enclen = (const int*)d_in[2];    // (B,)
    const int*   y      = (const int*)d_in[3];    // (B,L)
    const float* emb    = (const float*)d_in[4];  // (V,E)
    const float* W_ih   = (const float*)d_in[5];  // (4H, E+H)
    const float* W_hh   = (const float*)d_in[6];  // (4H, H)
    const float* b_ih   = (const float*)d_in[7];
    const float* b_hh   = (const float*)d_in[8];
    const float* b_out  = (const float*)d_in[9];

    float* pred     = (float*)d_out;                            // (B,L,V)
    float* attn_out = pred + (long long)BB * LL * VV;           // (L,T)

    // workspace layout (bytes), 256-aligned
    char* ws = (char*)d_ws;
    const size_t kv_elems = (size_t)BB * TT * HH;               // 24,576,000
    size_t off = 0;
    uint16_t* kbf = (uint16_t*)(ws + off); off += kv_elems * 2;                    // 49.15 MB
    uint16_t* vbf = (uint16_t*)(ws + off); off += kv_elems * 2;                    // 49.15 MB
    uint16_t* wpk = (uint16_t*)(ws + off); off += (size_t)NTW * KTW * 32 * 16 * 2; // 512 KB
    uint16_t* epk = (uint16_t*)(ws + off); off += (size_t)NTE * KTE * 32 * 16 * 2; // 504 KB
    off = (off + 255) & ~(size_t)255;
    float* h_st   = (float*)(ws + off); off += (size_t)BB * HH * 4;
    float* c_st   = (float*)(ws + off); off += (size_t)BB * HH * 4;
    float* ctx_st = (float*)(ws + off); off += (size_t)BB * HH * 4;
    (void)ws_size; (void)in_sizes; (void)n_in; (void)out_size;

    // one-time prep
    cvt_bf16_kernel<<<8192, 256, 0, stream>>>(key,   kbf, (long long)(kv_elems / 4));
    cvt_bf16_kernel<<<8192, 256, 0, stream>>>(value, vbf, (long long)(kv_elems / 4));
    pack_w_kernel  <<<512, 256, 0, stream>>>(W_ih, W_hh, wpk);
    pack_emb_kernel<<<512, 256, 0, stream>>>(emb, epk);
    init_state_kernel<<<(BB * HH + 255) / 256, 256, 0, stream>>>(value, h_st, c_st, ctx_st);

    // sequential decode
    for (int t = 0; t < LL; ++t) {
        lstm_gates_kernel<<<BB / 16, 512, 0, stream>>>(emb, y, wpk, b_ih, b_hh,
                                                       h_st, c_st, ctx_st, t);
        attention_kernel<<<BB, 256, 0, stream>>>(kbf, vbf, enclen, h_st, ctx_st,
                                                 attn_out, t);
        logits_kernel<<<dim3(8, 4), 256, 0, stream>>>(h_st, ctx_st, epk, b_out,
                                                      pred, t);
    }
}
